// GenieVideoTokenizer_43224550867218
// MI455X (gfx1250) — compile-verified, matrix operand order checked by reading, not run on hardware
//
#include <hip/hip_runtime.h>
#include <cstddef>
#include <cstdint>

// ---------------- problem constants (match reference) ----------------
constexpr int B_   = 2;
constexpr int T_   = 16;
constexpr int IH_  = 224;
constexpr int IW_  = 224;
constexpr int C_   = 3;
constexpr int TP_  = 4;
constexpr int PS_  = 16;
constexpr int D_   = 512;
constexpr int DFF_ = 2048;
constexpr int L_   = 6;
constexpr int NH_  = 8;
constexpr int K_   = 8192;
constexpr int Tp_  = T_ / TP_;             // 4
constexpr int Hp_  = IH_ / PS_;            // 14
constexpr int Wp_  = IW_ / PS_;            // 14
constexpr int S_   = Tp_ * Hp_ * Wp_;      // 784
constexpr int PIN_ = TP_ * PS_ * PS_ * C_; // 3072
constexpr int BS_  = B_ * S_;              // 1568
constexpr int HD_  = D_ / NH_;             // 64
constexpr int RECON_N = B_ * T_ * IH_ * IW_ * C_; // 4816896

typedef __attribute__((ext_vector_type(16))) _Float16 v16h;
typedef __attribute__((ext_vector_type(8)))  float    v8f;

// ---------------- small device helpers ----------------
__device__ __forceinline__ float gelu_tanh(float x) {
  float x3 = x * x * x;
  return 0.5f * x * (1.0f + tanhf(0.7978845608028654f * (x + 0.044715f * x3)));
}

// ---------------- fp32 -> fp16 convert (row-major copy) ----------------
__global__ __launch_bounds__(256) void cvt16_k(const float* __restrict__ x,
                                               _Float16* __restrict__ y, int n) {
  int i = blockIdx.x * 256 + threadIdx.x;
  if (i < n) y[i] = (_Float16)x[i];
}

// ---------------- fp32 [K,N] -> fp16 transposed [N,K] ----------------
__global__ __launch_bounds__(256) void cvtT16_k(const float* __restrict__ W,
                                                _Float16* __restrict__ WT,
                                                int Kd, int N) {
  int i = blockIdx.x * 256 + threadIdx.x;   // coalesced over W
  if (i >= Kd * N) return;
  int n = i % N, k = i / N;
  WT[(size_t)n * Kd + k] = (_Float16)W[i];
}

// ---------------- zero fill ----------------
__global__ __launch_bounds__(256) void zero_k(float* __restrict__ p, int n) {
  int i = blockIdx.x * 256 + threadIdx.x;
  if (i < n) p[i] = 0.0f;
}

// ---------------- WMMA GEMM ----------------
// C[M,N] = A[M,K] * B[K,N] (+bias)(+gelu)(+residual add)
// A fp16 row-major [M,K]; B supplied TRANSPOSED fp16 [N,K]; C fp32.
// 256 threads = 8 waves. Workgroup tile 128(M) x 64(N), K-step 32.
// Wave (wm=wave&3, wn=wave>>2) owns rows [wm*32,+32) x cols [wn*32,+32):
// 4 accumulators (2 A-frags x 2 B-frags) -> 4 WMMAs per K-step.
//
// LDS tiles are FRAGMENT-ORDERED: per 16x16 subtile, per lane, the 16 halves
// of its v16h fragment are contiguous (two 8-half runs of consecutive K, per
// the 16-bit A/B VGPR layout in CDNA5 ISA). Staging:
//   A: global_load_b128 (row clamped in-bounds, zero-selected) + ds_store_b128
//   B: GLOBAL_LOAD_ASYNC_TO_LDS_B128 (ASYNCcnt) - memory writes LDS directly,
//      fenced with s_wait_asynccnt 0 before the workgroup barrier.
template <int ACT, int HASADD>
__global__ __launch_bounds__(256) void gemm_wmma_k(
    const _Float16* __restrict__ A, const _Float16* __restrict__ BT,
    const float* __restrict__ bias, const float* __restrict__ add,
    float* __restrict__ C, int M, int N, int Kd) {
  // A: 8 row-subtiles x 32 lanes x 16 halves = 8KB
  // B: 4 col-subtiles x 32 lanes x 16 halves = 4KB
  __shared__ _Float16 AsF[8 * 32 * 16];
  __shared__ _Float16 BsF[4 * 32 * 16];

  const int tid  = threadIdx.x;
  const int wave = tid >> 5;
  const int lane = tid & 31;
  const int lo   = lane & 15;
  const int hi   = lane >> 4;
  const int wm   = wave & 3;   // row group (32 rows)
  const int wn   = wave >> 2;  // col group (32 cols)
  const int m0   = blockIdx.y * 128;
  const int n0   = blockIdx.x * 64;

  v8f c00 = {}, c01 = {}, c10 = {}, c11 = {};

  // Staging coordinates (fixed across K loop).
  // A: 512 chunks of 8 halves; 2 chunks per thread.
  const int ca0 = tid * 2;
  const int ca1 = tid * 2 + 1;
  const int a_s0 = ca0 >> 6,        a_s1 = ca1 >> 6;
  const int a_l0 = (ca0 >> 1) & 31, a_l1 = (ca1 >> 1) & 31;
  const int a_h0 = ca0 & 1,         a_h1 = ca1 & 1;
  const int a_row0 = m0 + a_s0 * 16 + (a_l0 & 15);
  const int a_row1 = m0 + a_s1 * 16 + (a_l1 & 15);
  const bool a_ok0 = a_row0 < M;
  const bool a_ok1 = a_row1 < M;
  const int a_rc0 = a_ok0 ? a_row0 : (M - 1);   // clamped: load always in-bounds
  const int a_rc1 = a_ok1 ? a_row1 : (M - 1);
  const int a_k0 = a_h0 * 16 + (a_l0 >> 4) * 8;
  const int a_k1 = a_h1 * 16 + (a_l1 >> 4) * 8;
  // B: 256 chunks; 1 chunk per thread.
  const int cb   = tid;
  const int b_s  = cb >> 6;
  const int b_l  = (cb >> 1) & 31;
  const int b_h  = cb & 1;
  const int b_col = n0 + b_s * 16 + (b_l & 15);  // N multiple of 64 -> in range
  const int b_k   = b_h * 16 + (b_l >> 4) * 8;

  const int4 zero4 = {0, 0, 0, 0};
#if defined(__gfx1250__)
  const unsigned b_lds = (unsigned)(uintptr_t)(BsF + cb * 8);
#endif

  for (int k0 = 0; k0 < Kd; k0 += 32) {
    if (k0 + 32 < Kd) {  // global_prefetch_b8 hints for next K tiles
      __builtin_prefetch(&A[(size_t)a_rc0 * Kd + k0 + 32 + a_k0], 0, 1);
      __builtin_prefetch(&BT[(size_t)b_col * Kd + k0 + 32 + b_k], 0, 1);
    }
    // ---- stage B: async global -> LDS (no VGPR round-trip) ----
#if defined(__gfx1250__)
    {
      unsigned long long ga =
          (unsigned long long)(uintptr_t)(BT + (size_t)b_col * Kd + k0 + b_k);
      asm volatile("global_load_async_to_lds_b128 %0, %1, off"
                   :: "v"(b_lds), "v"(ga) : "memory");
    }
#else
    *(int4*)(BsF + cb * 8) = *(const int4*)(BT + (size_t)b_col * Kd + k0 + b_k);
#endif
    // ---- stage A: unconditional b128 load (row clamped), zero-select ----
    {
      int4 va0 = *(const int4*)(A + (size_t)a_rc0 * Kd + k0 + a_k0);
      int4 va1 = *(const int4*)(A + (size_t)a_rc1 * Kd + k0 + a_k1);
      if (!a_ok0) va0 = zero4;
      if (!a_ok1) va1 = zero4;
      *(int4*)(AsF + ca0 * 8) = va0;
      *(int4*)(AsF + ca1 * 8) = va1;
    }
#if defined(__gfx1250__)
#if __has_builtin(__builtin_amdgcn_s_wait_asynccnt)
    __builtin_amdgcn_s_wait_asynccnt(0);
#else
    asm volatile("s_wait_asynccnt 0x0" ::: "memory");
#endif
#endif
    __syncthreads();

    // ---- fragments: one contiguous 32B LDS read each ----
    v16h a0 = *(const v16h*)(AsF + ((size_t)((2 * wm + 0) * 32 + lane)) * 16);
    v16h a1 = *(const v16h*)(AsF + ((size_t)((2 * wm + 1) * 32 + lane)) * 16);
    v16h b0 = *(const v16h*)(BsF + ((size_t)((2 * wn + 0) * 32 + lane)) * 16);
    v16h b1 = *(const v16h*)(BsF + ((size_t)((2 * wn + 1) * 32 + lane)) * 16);

    c00 = __builtin_amdgcn_wmma_f32_16x16x32_f16(false, a0, false, b0, (short)0, c00, false, false);
    c01 = __builtin_amdgcn_wmma_f32_16x16x32_f16(false, a0, false, b1, (short)0, c01, false, false);
    c10 = __builtin_amdgcn_wmma_f32_16x16x32_f16(false, a1, false, b0, (short)0, c10, false, false);
    c11 = __builtin_amdgcn_wmma_f32_16x16x32_f16(false, a1, false, b1, (short)0, c11, false, false);
    __syncthreads();
  }

  // ---- epilogue: C VGPR r -> row = r + hi*8, col = lo within 16x16 tile ----
#pragma unroll
  for (int ai = 0; ai < 2; ++ai) {
    const v8f* cr0 = ai ? &c10 : &c00;
    const v8f* cr1 = ai ? &c11 : &c01;
#pragma unroll
    for (int r = 0; r < 8; ++r) {
      int row = m0 + (2 * wm + ai) * 16 + hi * 8 + r;
      if (row >= M) continue;
      {
        int col = n0 + (2 * wn + 0) * 16 + lo;
        float v = (*cr0)[r] + bias[col];
        if (ACT) v = gelu_tanh(v);
        if (HASADD) v += add[(size_t)row * N + col];
        C[(size_t)row * N + col] = v;
      }
      {
        int col = n0 + (2 * wn + 1) * 16 + lo;
        float v = (*cr1)[r] + bias[col];
        if (ACT) v = gelu_tanh(v);
        if (HASADD) v += add[(size_t)row * N + col];
        C[(size_t)row * N + col] = v;
      }
    }
  }
}

// ---------------- LayerNorm over last dim ----------------
__global__ __launch_bounds__(256) void layernorm_k(
    const float* __restrict__ x, const float* __restrict__ s,
    const float* __restrict__ b, float* __restrict__ y, int dim) {
  __shared__ float red[256];
  const int row = blockIdx.x;
  const int tid = threadIdx.x;
  const float* xr = x + (size_t)row * dim;
  float sum = 0.0f;
  for (int i = tid; i < dim; i += 256) sum += xr[i];
  red[tid] = sum; __syncthreads();
  for (int st = 128; st > 0; st >>= 1) { if (tid < st) red[tid] += red[tid + st]; __syncthreads(); }
  float mean = red[0] / dim;
  __syncthreads();
  float vs = 0.0f;
  for (int i = tid; i < dim; i += 256) { float d = xr[i] - mean; vs += d * d; }
  red[tid] = vs; __syncthreads();
  for (int st = 128; st > 0; st >>= 1) { if (tid < st) red[tid] += red[tid + st]; __syncthreads(); }
  float inv = rsqrtf(red[0] / dim + 1e-6f);
  float* yr = y + (size_t)row * dim;
  for (int i = tid; i < dim; i += 256) yr[i] = (xr[i] - mean) * inv * s[i] + b[i];
}

// ---------------- flash attention: one wave32 per (b,h,query) ----------------
__global__ __launch_bounds__(32) void attn_flash_k(
    const float* __restrict__ q, const float* __restrict__ k,
    const float* __restrict__ v, float* __restrict__ o) {
  const int id   = blockIdx.x;             // b*NH*S + h*S + qi
  const int qi   = id % S_;
  const int h    = (id / S_) % NH_;
  const int b    = id / (S_ * NH_);
  const int lane = threadIdx.x;
  const float scale = 0.125f;              // 1/sqrt(64)
  const size_t rowq = (size_t)(b * S_ + qi) * D_ + h * HD_;
  const float q0 = q[rowq + lane];
  const float q1 = q[rowq + 32 + lane];
  float m = -1e30f, l = 0.0f, a0 = 0.0f, a1 = 0.0f;
  for (int s = 0; s < S_; ++s) {
    size_t rk = (size_t)(b * S_ + s) * D_ + h * HD_;
    float p = q0 * k[rk + lane] + q1 * k[rk + 32 + lane];
#pragma unroll
    for (int off = 16; off > 0; off >>= 1) p += __shfl_xor(p, off, 32);
    p *= scale;
    float mn   = fmaxf(m, p);
    float corr = __expf(m - mn);
    float w    = __expf(p - mn);
    l  = l * corr + w;
    a0 = a0 * corr + w * v[rk + lane];
    a1 = a1 * corr + w * v[rk + 32 + lane];
    m = mn;
  }
  o[rowq + lane]      = a0 / l;
  o[rowq + 32 + lane] = a1 / l;
}

// ---------------- positional embedding add ----------------
__global__ __launch_bounds__(256) void posadd_k(
    float* __restrict__ z, const float* __restrict__ pt,
    const float* __restrict__ ph, const float* __restrict__ pw, int n) {
  int i = blockIdx.x * 256 + threadIdx.x;
  if (i >= n) return;
  int d  = i % D_;
  int s  = (i / D_) % S_;
  int t  = s / (Hp_ * Wp_);
  int hw = s % (Hp_ * Wp_);
  int h  = hw / Wp_;
  int w  = hw % Wp_;
  z[i] += pt[t * D_ + d] + ph[h * D_ + d] + pw[w * D_ + d];
}

// ---------------- 3D patchify / unpatchify ----------------
__global__ __launch_bounds__(256) void patchify_k(const float* __restrict__ video,
                                                  float* __restrict__ out, int n) {
  int i = blockIdx.x * 256 + threadIdx.x;
  if (i >= n) return;
  int pin = i % PIN_; int rest = i / PIN_;
  int wp = rest % Wp_; rest /= Wp_;
  int hp = rest % Hp_; rest /= Hp_;
  int tp = rest % Tp_; int b = rest / Tp_;
  int c  = pin % C_;   int p2 = pin / C_;
  int pw = p2 % PS_;   p2 /= PS_;
  int ph = p2 % PS_;   int tt = p2 / PS_;
  int t = tp * TP_ + tt, ih = hp * PS_ + ph, iw = wp * PS_ + pw;
  out[i] = video[((((size_t)b * T_ + t) * IH_ + ih) * IW_ + iw) * C_ + c];
}

__global__ __launch_bounds__(256) void unpatchify_k(const float* __restrict__ px,
                                                    float* __restrict__ recon, int n) {
  int i = blockIdx.x * 256 + threadIdx.x;
  if (i >= n) return;
  int c  = i % C_;  int rest = i / C_;
  int iw = rest % IW_; rest /= IW_;
  int ih = rest % IH_; rest /= IH_;
  int t  = rest % T_;  int b = rest / T_;
  int tp = t / TP_, tt = t % TP_;
  int hp = ih / PS_, ph = ih % PS_;
  int wp = iw / PS_, pw = iw % PS_;
  int pin = ((tt * PS_ + ph) * PS_ + pw) * C_ + c;
  recon[i] = px[(((((size_t)b * Tp_ + tp) * Hp_ + hp) * Wp_ + wp) * (size_t)PIN_) + pin];
}

// ---------------- VQ: fused argmin over codebook (one block per token) ----------------
__global__ __launch_bounds__(256) void vq_argmin_k(
    const float* __restrict__ ze, const float* __restrict__ cb,
    int* __restrict__ idx, float* __restrict__ counts) {
  __shared__ float zrow[D_];
  __shared__ float bd[256];
  __shared__ int   bi[256];
  const int row = blockIdx.x;
  const int tid = threadIdx.x;
  for (int i = tid; i < D_; i += 256) zrow[i] = ze[(size_t)row * D_ + i];
  __syncthreads();
  float best = 3.4e38f; int bestj = 0;
  for (int j = tid; j < K_; j += 256) {
    const float* c = cb + (size_t)j * D_;
    float d = 0.0f;
    for (int t = 0; t < D_; ++t) { float df = zrow[t] - c[t]; d += df * df; }
    if (d < best) { best = d; bestj = j; }
  }
  bd[tid] = best; bi[tid] = bestj; __syncthreads();
  for (int st = 128; st > 0; st >>= 1) {
    if (tid < st) {
      if (bd[tid + st] < bd[tid] ||
          (bd[tid + st] == bd[tid] && bi[tid + st] < bi[tid])) {
        bd[tid] = bd[tid + st]; bi[tid] = bi[tid + st];
      }
    }
    __syncthreads();
  }
  if (tid == 0) {
    idx[row] = bi[0];
    atomicAdd(&counts[bi[0]], 1.0f);
  }
}

// ---------------- z_q gather + commit-loss partial sums ----------------
__global__ __launch_bounds__(256) void vq_gather_loss_k(
    const float* __restrict__ ze, const float* __restrict__ cb,
    const int* __restrict__ idx, float* __restrict__ zq,
    float* __restrict__ acc, int n) {
  __shared__ float red[256];
  int i = blockIdx.x * 256 + threadIdx.x;
  float d2 = 0.0f;
  if (i < n) {
    int row = i / D_, col = i % D_;
    float qv = cb[(size_t)idx[row] * D_ + col];
    zq[i] = qv;
    float df = qv - ze[i];
    d2 = df * df;
  }
  red[threadIdx.x] = d2; __syncthreads();
  for (int st = 128; st > 0; st >>= 1) { if (threadIdx.x < st) red[threadIdx.x] += red[threadIdx.x + st]; __syncthreads(); }
  if (threadIdx.x == 0) atomicAdd(acc, red[0]);
}

// ---------------- perplexity entropy over codebook usage ----------------
__global__ __launch_bounds__(256) void entropy_k(const float* __restrict__ counts,
                                                 float* __restrict__ ent) {
  __shared__ float red[256];
  int j = blockIdx.x * 256 + threadIdx.x;
  float e = 0.0f;
  if (j < K_) {
    float p = counts[j] / (float)BS_;
    e = -p * __logf(p + 1e-10f);
  }
  red[threadIdx.x] = e; __syncthreads();
  for (int st = 128; st > 0; st >>= 1) { if (threadIdx.x < st) red[threadIdx.x] += red[threadIdx.x + st]; __syncthreads(); }
  if (threadIdx.x == 0) atomicAdd(ent, red[0]);
}

__global__ void idx_to_float_k(const int* __restrict__ idx, float* __restrict__ out, int n) {
  int i = blockIdx.x * 256 + threadIdx.x;
  if (i < n) out[i] = (float)idx[i];
}

__global__ void finalize_k(const float* __restrict__ acc, const float* __restrict__ ent,
                           float* __restrict__ scal) {
  if (threadIdx.x == 0 && blockIdx.x == 0) {
    float m = acc[0] / (float)((size_t)BS_ * D_);
    scal[0] = m;                // commit
    scal[1] = m;                // cb_loss (forward-equal)
    scal[2] = __expf(ent[0]);   // perplexity
  }
}

// =====================================================================
extern "C" void kernel_launch(void* const* d_in, const int* in_sizes, int n_in,
                              void* d_out, int out_size, void* d_ws, size_t ws_size,
                              hipStream_t stream) {
  (void)in_sizes; (void)n_in; (void)out_size; (void)ws_size;

  // -------- inputs (setup_inputs order) --------
  int a = 0;
  const float* video    = (const float*)d_in[a++];
  const float* e_patch_W = (const float*)d_in[a++];
  const float* e_patch_b = (const float*)d_in[a++];
  const float* e_pos_t  = (const float*)d_in[a++];
  const float* e_pos_h  = (const float*)d_in[a++];
  const float* e_pos_w  = (const float*)d_in[a++];
  const float* e_ln1_s  = (const float*)d_in[a++];
  const float* e_ln1_b  = (const float*)d_in[a++];
  const float* e_Wqkv   = (const float*)d_in[a++];
  const float* e_bqkv   = (const float*)d_in[a++];
  const float* e_Wo     = (const float*)d_in[a++];
  const float* e_bo     = (const float*)d_in[a++];
  const float* e_ln2_s  = (const float*)d_in[a++];
  const float* e_ln2_b  = (const float*)d_in[a++];
  const float* e_W1     = (const float*)d_in[a++];
  const float* e_b1     = (const float*)d_in[a++];
  const float* e_W2     = (const float*)d_in[a++];
  const float* e_b2     = (const float*)d_in[a++];
  const float* e_lnf_s  = (const float*)d_in[a++];
  const float* e_lnf_b  = (const float*)d_in[a++];
  const float* e_cb_W   = (const float*)d_in[a++];
  const float* e_cb_b   = (const float*)d_in[a++];
  const float* codebook = (const float*)d_in[a++];
  const float* d_cb_W   = (const float*)d_in[a++];
  const float* d_cb_b   = (const float*)d_in[a++];
  const float* d_pos_t  = (const float*)d_in[a++];
  const float* d_pos_h  = (const float*)d_in[a++];
  const float* d_pos_w  = (const float*)d_in[a++];
  const float* d_ln1_s  = (const float*)d_in[a++];
  const float* d_ln1_b  = (const float*)d_in[a++];
  const float* d_Wqkv   = (const float*)d_in[a++];
  const float* d_bqkv   = (const float*)d_in[a++];
  const float* d_Wo     = (const float*)d_in[a++];
  const float* d_bo     = (const float*)d_in[a++];
  const float* d_ln2_s  = (const float*)d_in[a++];
  const float* d_ln2_b  = (const float*)d_in[a++];
  const float* d_W1     = (const float*)d_in[a++];
  const float* d_b1     = (const float*)d_in[a++];
  const float* d_W2     = (const float*)d_in[a++];
  const float* d_b2     = (const float*)d_in[a++];
  const float* d_lnf_s  = (const float*)d_in[a++];
  const float* d_lnf_b  = (const float*)d_in[a++];
  const float* d_px_W   = (const float*)d_in[a++];
  const float* d_px_b   = (const float*)d_in[a++];

  // -------- workspace bump allocator --------
  char* wsp = (char*)d_ws;
  auto alloc = [&](size_t bytes) -> void* {
    void* p = (void*)wsp;
    wsp += (bytes + 255) & ~(size_t)255;
    return p;
  };
  float*    patchv = (float*)alloc((size_t)BS_ * PIN_ * 4); // also pixel-head output
  float*    xA     = (float*)alloc((size_t)BS_ * D_ * 4);
  float*    xB     = (float*)alloc((size_t)BS_ * D_ * 4);
  float*    xn     = (float*)alloc((size_t)BS_ * D_ * 4);
  float*    qkv    = (float*)alloc((size_t)3 * BS_ * D_ * 4);
  float*    hbuf   = (float*)alloc((size_t)BS_ * DFF_ * 4);
  float*    ze     = (float*)alloc((size_t)BS_ * D_ * 4);
  float*    zq     = (float*)alloc((size_t)BS_ * D_ * 4);
  _Float16* a16    = (_Float16*)alloc((size_t)BS_ * PIN_ * 2);
  _Float16* b16    = (_Float16*)alloc((size_t)PIN_ * D_ * 2);
  int*      idxb   = (int*)alloc((size_t)BS_ * 4);
  float*    counts = (float*)alloc((size_t)K_ * 4);
  float*    acc    = (float*)alloc(16 * 4);  // [0]=commit accum, [1]=entropy accum

  // -------- launch helpers --------
  auto cvt = [&](const float* src, _Float16* dst, int n) {
    cvt16_k<<<(n + 255) / 256, 256, 0, stream>>>(src, dst, n);
  };
  auto cvtT = [&](const float* W, _Float16* WT, int Kd, int N) {
    cvtT16_k<<<(Kd * N + 255) / 256, 256, 0, stream>>>(W, WT, Kd, N);
  };
  auto gemm = [&](const _Float16* A, const _Float16* BT, const float* bias,
                  const float* add, float* Cc, int M, int N, int Kd, int act) {
    dim3 grid(N / 64, (M + 127) / 128);
    if (add) {
      if (act) gemm_wmma_k<1, 1><<<grid, 256, 0, stream>>>(A, BT, bias, add, Cc, M, N, Kd);
      else     gemm_wmma_k<0, 1><<<grid, 256, 0, stream>>>(A, BT, bias, add, Cc, M, N, Kd);
    } else {
      if (act) gemm_wmma_k<1, 0><<<grid, 256, 0, stream>>>(A, BT, bias, nullptr, Cc, M, N, Kd);
      else     gemm_wmma_k<0, 0><<<grid, 256, 0, stream>>>(A, BT, bias, nullptr, Cc, M, N, Kd);
    }
  };

  struct StackParams {
    const float *ln1_s, *ln1_b, *Wqkv, *bqkv, *Wo, *bo;
    const float *ln2_s, *ln2_b, *W1, *b1, *W2, *b2;
  };
  auto run_stack = [&](float*& cur, float*& alt, const StackParams& P) {
    for (int l = 0; l < L_; ++l) {
      // ---- attention sublayer ----
      layernorm_k<<<BS_, 256, 0, stream>>>(cur, P.ln1_s + l * D_, P.ln1_b + l * D_, xn, D_);
      cvt(xn, a16, BS_ * D_);
      for (int wsel = 0; wsel < 3; ++wsel) {
        cvtT(P.Wqkv + ((size_t)l * 3 + wsel) * D_ * D_, b16, D_, D_);
        gemm(a16, b16, P.bqkv + (l * 3 + wsel) * D_, nullptr,
             qkv + (size_t)wsel * BS_ * D_, BS_, D_, D_, 0);
      }
      attn_flash_k<<<B_ * NH_ * S_, 32, 0, stream>>>(
          qkv, qkv + (size_t)BS_ * D_, qkv + (size_t)2 * BS_ * D_, xn);
      cvt(xn, a16, BS_ * D_);
      cvtT(P.Wo + (size_t)l * D_ * D_, b16, D_, D_);
      gemm(a16, b16, P.bo + l * D_, cur, alt, BS_, D_, D_, 0);
      { float* t = cur; cur = alt; alt = t; }
      // ---- MLP sublayer ----
      layernorm_k<<<BS_, 256, 0, stream>>>(cur, P.ln2_s + l * D_, P.ln2_b + l * D_, xn, D_);
      cvt(xn, a16, BS_ * D_);
      cvtT(P.W1 + (size_t)l * D_ * DFF_, b16, D_, DFF_);
      gemm(a16, b16, P.b1 + (size_t)l * DFF_, nullptr, hbuf, BS_, DFF_, D_, 1);
      cvt(hbuf, a16, BS_ * DFF_);
      cvtT(P.W2 + (size_t)l * DFF_ * D_, b16, DFF_, D_);
      gemm(a16, b16, P.b2 + l * D_, cur, alt, BS_, D_, DFF_, 0);
      { float* t = cur; cur = alt; alt = t; }
    }
  };

  float* out    = (float*)d_out;
  float* reconO = out;
  float* idxO   = out + RECON_N;
  float* scalO  = out + RECON_N + BS_;

  // -------- per-call state reset (graph-replay safe) --------
  zero_k<<<(K_ + 255) / 256, 256, 0, stream>>>(counts, K_);
  zero_k<<<1, 256, 0, stream>>>(acc, 16);

  // ================= encoder =================
  patchify_k<<<((size_t)BS_ * PIN_ + 255) / 256, 256, 0, stream>>>(video, patchv, BS_ * PIN_);
  cvt(patchv, a16, BS_ * PIN_);
  cvtT(e_patch_W, b16, PIN_, D_);
  float* cur = xA; float* alt = xB;
  gemm(a16, b16, e_patch_b, nullptr, cur, BS_, D_, PIN_, 0);
  posadd_k<<<(BS_ * D_ + 255) / 256, 256, 0, stream>>>(cur, e_pos_t, e_pos_h, e_pos_w, BS_ * D_);

  StackParams EP{e_ln1_s, e_ln1_b, e_Wqkv, e_bqkv, e_Wo, e_bo,
                 e_ln2_s, e_ln2_b, e_W1, e_b1, e_W2, e_b2};
  run_stack(cur, alt, EP);

  layernorm_k<<<BS_, 256, 0, stream>>>(cur, e_lnf_s, e_lnf_b, xn, D_);
  cvt(xn, a16, BS_ * D_);
  cvtT(e_cb_W, b16, D_, D_);
  gemm(a16, b16, e_cb_b, nullptr, ze, BS_, D_, D_, 0);

  // ================= VQ =================
  vq_argmin_k<<<BS_, 256, 0, stream>>>(ze, codebook, idxb, counts);
  vq_gather_loss_k<<<(BS_ * D_ + 255) / 256, 256, 0, stream>>>(
      ze, codebook, idxb, zq, &acc[0], BS_ * D_);
  entropy_k<<<K_ / 256, 256, 0, stream>>>(counts, &acc[1]);
  idx_to_float_k<<<(BS_ + 255) / 256, 256, 0, stream>>>(idxb, idxO, BS_);
  finalize_k<<<1, 32, 0, stream>>>(&acc[0], &acc[1], scalO);

  // ================= decoder =================
  cvt(zq, a16, BS_ * D_);
  cvtT(d_cb_W, b16, D_, D_);
  cur = xA; alt = xB;
  gemm(a16, b16, d_cb_b, nullptr, cur, BS_, D_, D_, 0);
  posadd_k<<<(BS_ * D_ + 255) / 256, 256, 0, stream>>>(cur, d_pos_t, d_pos_h, d_pos_w, BS_ * D_);

  StackParams DP{d_ln1_s, d_ln1_b, d_Wqkv, d_bqkv, d_Wo, d_bo,
                 d_ln2_s, d_ln2_b, d_W1, d_b1, d_W2, d_b2};
  run_stack(cur, alt, DP);

  layernorm_k<<<BS_, 256, 0, stream>>>(cur, d_lnf_s, d_lnf_b, xn, D_);
  cvt(xn, a16, BS_ * D_);
  cvtT(d_px_W, b16, D_, PIN_);
  gemm(a16, b16, d_px_b, nullptr, patchv, BS_, PIN_, D_, 0);   // reuse patch buffer
  unpatchify_k<<<(RECON_N + 255) / 256, 256, 0, stream>>>(patchv, reconO, RECON_N);
}
